// QuantumConv2d_12524124635734
// MI455X (gfx1250) — compile-verified, matrix operand order checked by reading, not run on hardware
//
#include <hip/hip_runtime.h>
#include <hip/hip_bf16.h>
#include <math.h>

typedef __attribute__((ext_vector_type(16))) _Float16 v16h;
typedef __attribute__((ext_vector_type(8)))  float    v8f;

#define PATCH 288
#define QD 64
#define C_IN 32
#define ROWSTRIDE_A 296   // halfs; 592 B rows, 16B aligned
#define ROWSTRIDE_Y 72    // halfs; 144 B rows, 16B aligned

// workspace layout (in _Float16 units)
#define WS_W   0                  // 4 nt * 9 kt * 32 lanes * 16 halfs = 18432
#define WS_MR  18432              // 4 nt * 2 kt * 512 = 4096
#define WS_MI  (18432 + 4096)
#define WS_P   (18432 + 8192)
// total 30720 halfs = 61440 bytes of d_ws

union FragU { uint4 u[2]; v16h v; };

// ---------------------------------------------------------------------------
// Prep kernel: compose the full quantum circuit into a 64x64 complex matrix
// (M_r, M_i) and pre-swizzle all GEMM B-matrices into WMMA fragment order.
// ---------------------------------------------------------------------------
__global__ void __launch_bounds__(256)
qprep_kernel(const float* __restrict__ in_proj_w,
             const float* __restrict__ weights,
             const float* __restrict__ meas_w,
             const float* __restrict__ out_proj_w,
             _Float16* __restrict__ ws)
{
    __shared__ float Mr[64 * 64];
    __shared__ float Mi[64 * 64];
    const int tid = threadIdx.x;

    for (int e = tid; e < 4096; e += 256) {
        int r = e >> 6, c = e & 63;
        Mr[e] = (r == c) ? 1.f : 0.f;
        Mi[e] = 0.f;
    }
    __syncthreads();

    // Apply 2x2 complex gate on wire w (wire 0 = MSB -> bit position 5-w)
    auto apply1q = [&](int wire, float u00r, float u00i, float u01r, float u01i,
                       float u10r, float u10i, float u11r, float u11i) {
        int p = 5 - wire;
        for (int e = tid; e < 2048; e += 256) {
            int pair = e >> 6, col = e & 63;
            int low = pair & ((1 << p) - 1);
            int i0 = ((pair >> p) << (p + 1)) | low;
            int i1 = i0 | (1 << p);
            float ar = Mr[i0 * 64 + col], ai = Mi[i0 * 64 + col];
            float br = Mr[i1 * 64 + col], bi = Mi[i1 * 64 + col];
            Mr[i0 * 64 + col] = u00r * ar - u00i * ai + u01r * br - u01i * bi;
            Mi[i0 * 64 + col] = u00r * ai + u00i * ar + u01r * bi + u01i * br;
            Mr[i1 * 64 + col] = u10r * ar - u10i * ai + u11r * br - u11i * bi;
            Mi[i1 * 64 + col] = u10r * ai + u10i * ar + u11r * bi + u11i * br;
        }
        __syncthreads();
    };
    // CNOT control=wire, target=wire+1 -> bits pc=5-wire, pt=pc-1
    auto applyCnot = [&](int wire) {
        int pc = 5 - wire, pt = pc - 1;
        for (int e = tid; e < 1024; e += 256) {
            int grp = e >> 6, col = e & 63;
            int low = grp & ((1 << pt) - 1);
            int high = grp >> pt;
            int i0 = (high << (pc + 1)) | (1 << pc) | low;   // control=1, target=0
            int i1 = i0 | (1 << pt);
            float tr = Mr[i0 * 64 + col], ti = Mi[i0 * 64 + col];
            Mr[i0 * 64 + col] = Mr[i1 * 64 + col];
            Mi[i0 * 64 + col] = Mi[i1 * 64 + col];
            Mr[i1 * 64 + col] = tr;
            Mi[i1 * 64 + col] = ti;
        }
        __syncthreads();
    };

    for (int l = 0; l < 2; ++l) {
        for (int i = 0; i < 6; ++i) {
            float t = weights[(l * 6 + i) * 3 + 0];
            float c = cosf(t * 0.5f), s = sinf(t * 0.5f);
            apply1q(i, c, 0.f, 0.f, -s, 0.f, -s, c, 0.f);      // RX
            t = weights[(l * 6 + i) * 3 + 1];
            c = cosf(t * 0.5f); s = sinf(t * 0.5f);
            apply1q(i, c, 0.f, -s, 0.f, s, 0.f, c, 0.f);       // RY
        }
        for (int i = 0; i < 5; ++i) applyCnot(i);
        for (int i = 0; i < 6; ++i) {
            float t = weights[(l * 6 + i) * 3 + 2];
            float c = cosf(t * 0.5f), s = sinf(t * 0.5f);
            apply1q(i, c, 0.f, -s, 0.f, s, 0.f, c, 0.f);       // RY
        }
    }
    for (int i = 0; i < 6; ++i) {
        float t = meas_w[i * 3 + 0], p = meas_w[i * 3 + 1], lm = meas_w[i * 3 + 2];
        float c = cosf(t * 0.5f), s = sinf(t * 0.5f);
        float cp = cosf(p), sp = sinf(p);
        float cl = cosf(lm), sl = sinf(lm);
        float cpl = cosf(p + lm), spl = sinf(p + lm);
        apply1q(i, c, 0.f, -cl * s, -sl * s, cp * s, sp * s, cpl * c, spl * c); // U3
    }

    // Swizzle in_proj (K=288, N=64): B[k][n] = in_proj_w[n*288+k]
    for (int e = tid; e < 4 * 9 * 512; e += 256) {
        int h = e & 15, lane = (e >> 4) & 31, s = e >> 9;
        int kt = s % 9, nt = s / 9;
        int n = nt * 16 + (lane & 15);
        int base = (lane & 16) ? 8 : 0;
        int kl = (h < 8) ? (base + h) : (base + h + 8);
        int k = kt * 32 + kl;
        ws[WS_W + e] = (_Float16)in_proj_w[n * PATCH + k];
    }
    // Swizzle Mr, Mi, out_proj (all K=64, N=64)
    for (int e = tid; e < 4 * 2 * 512; e += 256) {
        int h = e & 15, lane = (e >> 4) & 31, s = e >> 9;
        int kt = s & 1, nt = s >> 1;
        int n = nt * 16 + (lane & 15);
        int base = (lane & 16) ? 8 : 0;
        int kl = (h < 8) ? (base + h) : (base + h + 8);
        int k = kt * 32 + kl;
        ws[WS_MR + e] = (_Float16)Mr[n * 64 + k];
        ws[WS_MI + e] = (_Float16)Mi[n * 64 + k];
        ws[WS_P + e]  = (_Float16)out_proj_w[n * 64 + k];
    }
}

// ---------------------------------------------------------------------------
// Fused main kernel: 64-pixel tile per workgroup, 8 waves.
// wave wv: m-tile = wv>>1 (16 rows), n-half = wv&1 (32 of 64 cols)
// ---------------------------------------------------------------------------
__global__ void __launch_bounds__(256)
qconv_main_kernel(const float* __restrict__ x,
                  const float* __restrict__ inp_scale,
                  const float* __restrict__ in_proj_b,
                  const float* __restrict__ out_proj_b,
                  const _Float16* __restrict__ ws,
                  float* __restrict__ out)
{
    __shared__ __align__(16) _Float16 bufA[64 * ROWSTRIDE_A]; // patches; reused as f32 outstage[64][65]
    __shared__ __align__(16) _Float16 bufY[64 * ROWSTRIDE_Y]; // normalized state
    __shared__ __align__(16) _Float16 bufP[64 * ROWSTRIDE_Y]; // probabilities
    __shared__ float rowsum[64][2];

    const int tid  = threadIdx.x;
    const int lane = tid & 31;
    const int wv   = tid >> 5;
    const int mt   = wv >> 1;
    const int nh   = wv & 1;
    const int pix0 = blockIdx.x * 64;

    // ---- stage 64x288 patch tile into LDS (f16, scaled) ----
    {
        const int m = tid & 63;
        const int pixel = pix0 + m;
        const int b  = pixel >> 12;
        const int hw = pixel & 4095;
        const int hh0 = (hw >> 6) - 1;
        const int ww0 = (hw & 63) - 1;
        for (int k = tid >> 6; k < PATCH; k += 4) {
            int c = k / 9;
            int rem = k - c * 9;
            int r = rem / 3;
            int s2 = rem - r * 3;
            int hh = hh0 + r, wwp = ww0 + s2;
            float v = 0.f;
            if ((unsigned)hh < 64u && (unsigned)wwp < 64u)
                v = x[(((size_t)b * C_IN + c) * 64 + hh) * 64 + wwp];
            bufA[m * ROWSTRIDE_A + k] = (_Float16)(v * inp_scale[k]);
        }
    }
    __syncthreads();

    const int nlo  = lane & 15;            // A-row within tile / C column
    const int base = (lane & 16) ? 8 : 0;  // K sub-offset (A frag) / C row offset

    // ---- GEMM1: y = patch @ in_proj_w^T ----
    v8f acc0 = {}, acc1 = {};
    {
        const _Float16* Arow = &bufA[(mt * 16 + nlo) * ROWSTRIDE_A];
        for (int kt = 0; kt < 9; ++kt) {
            FragU af, bf0, bf1;
            const _Float16* pa = Arow + kt * 32 + base;
            af.u[0] = *(const uint4*)(pa);
            af.u[1] = *(const uint4*)(pa + 16);
            const _Float16* pb0 = ws + WS_W + (((2 * nh + 0) * 9 + kt) * 512 + lane * 16);
            const _Float16* pb1 = ws + WS_W + (((2 * nh + 1) * 9 + kt) * 512 + lane * 16);
            bf0.u[0] = *(const uint4*)(pb0); bf0.u[1] = *(const uint4*)(pb0 + 8);
            bf1.u[0] = *(const uint4*)(pb1); bf1.u[1] = *(const uint4*)(pb1 + 8);
            acc0 = __builtin_amdgcn_wmma_f32_16x16x32_f16(false, af.v, false, bf0.v, (short)0, acc0, false, false);
            acc1 = __builtin_amdgcn_wmma_f32_16x16x32_f16(false, af.v, false, bf1.v, (short)0, acc1, false, false);
        }
    }

    // ---- bias + row L2 norm (shfl within 16-lane groups + cross-wave LDS) ----
    const int n0 = (2 * nh + 0) * 16 + nlo;
    const int n1 = (2 * nh + 1) * 16 + nlo;
    float b0 = in_proj_b[n0], b1 = in_proj_b[n1];
    float y0[8], y1[8], part[8];
    for (int r = 0; r < 8; ++r) {
        y0[r] = acc0[r] + b0;
        y1[r] = acc1[r] + b1;
        part[r] = y0[r] * y0[r] + y1[r] * y1[r];
    }
    for (int mask = 1; mask <= 8; mask <<= 1)
        for (int r = 0; r < 8; ++r)
            part[r] += __shfl_xor(part[r], mask, 32);
    if (nlo == 0) {
        for (int r = 0; r < 8; ++r)
            rowsum[mt * 16 + base + r][nh] = part[r];
    }
    __syncthreads();
    for (int r = 0; r < 8; ++r) {
        int m = mt * 16 + base + r;
        float tot = rowsum[m][0] + rowsum[m][1];
        float inv = 1.f / (sqrtf(tot) + 1e-9f);
        bufY[m * ROWSTRIDE_Y + n0] = (_Float16)(y0[r] * inv);
        bufY[m * ROWSTRIDE_Y + n1] = (_Float16)(y1[r] * inv);
    }
    __syncthreads();

    // ---- GEMM2: a = y_n @ M_r^T, b = y_n @ M_i^T ----
    v8f ar0 = {}, ar1 = {}, ai0 = {}, ai1 = {};
    {
        const _Float16* Yrow = &bufY[(mt * 16 + nlo) * ROWSTRIDE_Y];
        for (int kt = 0; kt < 2; ++kt) {
            FragU af, br0, br1, bi0, bi1;
            const _Float16* pa = Yrow + kt * 32 + base;
            af.u[0] = *(const uint4*)(pa);
            af.u[1] = *(const uint4*)(pa + 16);
            const _Float16* q;
            q = ws + WS_MR + (((2 * nh + 0) * 2 + kt) * 512 + lane * 16);
            br0.u[0] = *(const uint4*)(q); br0.u[1] = *(const uint4*)(q + 8);
            q = ws + WS_MR + (((2 * nh + 1) * 2 + kt) * 512 + lane * 16);
            br1.u[0] = *(const uint4*)(q); br1.u[1] = *(const uint4*)(q + 8);
            q = ws + WS_MI + (((2 * nh + 0) * 2 + kt) * 512 + lane * 16);
            bi0.u[0] = *(const uint4*)(q); bi0.u[1] = *(const uint4*)(q + 8);
            q = ws + WS_MI + (((2 * nh + 1) * 2 + kt) * 512 + lane * 16);
            bi1.u[0] = *(const uint4*)(q); bi1.u[1] = *(const uint4*)(q + 8);
            ar0 = __builtin_amdgcn_wmma_f32_16x16x32_f16(false, af.v, false, br0.v, (short)0, ar0, false, false);
            ar1 = __builtin_amdgcn_wmma_f32_16x16x32_f16(false, af.v, false, br1.v, (short)0, ar1, false, false);
            ai0 = __builtin_amdgcn_wmma_f32_16x16x32_f16(false, af.v, false, bi0.v, (short)0, ai0, false, false);
            ai1 = __builtin_amdgcn_wmma_f32_16x16x32_f16(false, af.v, false, bi1.v, (short)0, ai1, false, false);
        }
    }

    // ---- probs = a^2 + b^2 ----
    for (int r = 0; r < 8; ++r) {
        int m = mt * 16 + base + r;
        bufP[m * ROWSTRIDE_Y + n0] = (_Float16)(ar0[r] * ar0[r] + ai0[r] * ai0[r]);
        bufP[m * ROWSTRIDE_Y + n1] = (_Float16)(ar1[r] * ar1[r] + ai1[r] * ai1[r]);
    }
    __syncthreads();

    // ---- GEMM3: out = probs @ out_proj_w^T ----
    v8f o0 = {}, o1 = {};
    {
        const _Float16* Prow = &bufP[(mt * 16 + nlo) * ROWSTRIDE_Y];
        for (int kt = 0; kt < 2; ++kt) {
            FragU af, bp0, bp1;
            const _Float16* pa = Prow + kt * 32 + base;
            af.u[0] = *(const uint4*)(pa);
            af.u[1] = *(const uint4*)(pa + 16);
            const _Float16* q;
            q = ws + WS_P + (((2 * nh + 0) * 2 + kt) * 512 + lane * 16);
            bp0.u[0] = *(const uint4*)(q); bp0.u[1] = *(const uint4*)(q + 8);
            q = ws + WS_P + (((2 * nh + 1) * 2 + kt) * 512 + lane * 16);
            bp1.u[0] = *(const uint4*)(q); bp1.u[1] = *(const uint4*)(q + 8);
            o0 = __builtin_amdgcn_wmma_f32_16x16x32_f16(false, af.v, false, bp0.v, (short)0, o0, false, false);
            o1 = __builtin_amdgcn_wmma_f32_16x16x32_f16(false, af.v, false, bp1.v, (short)0, o1, false, false);
        }
    }

    // ---- stage into LDS (padded transpose), then coalesced NCHW store ----
    float ob0 = out_proj_b[n0], ob1 = out_proj_b[n1];
    float* outstage = (float*)bufA;   // 64*65*4 = 16640 B << bufA size
    for (int r = 0; r < 8; ++r) {
        int m = mt * 16 + base + r;
        outstage[m * 65 + n0] = o0[r] + ob0;
        outstage[m * 65 + n1] = o1[r] + ob1;
    }
    __syncthreads();
    for (int e = tid; e < 64 * 64; e += 256) {
        int co = e >> 6, m = e & 63;
        int pixel = pix0 + m;
        int b = pixel >> 12, hw = pixel & 4095;
        out[((size_t)(b * 64 + co)) * 4096 + hw] = outstage[m * 65 + co];
    }
}

extern "C" void kernel_launch(void* const* d_in, const int* in_sizes, int n_in,
                              void* d_out, int out_size, void* d_ws, size_t ws_size,
                              hipStream_t stream) {
    const float* x          = (const float*)d_in[0];
    const float* inp_scale  = (const float*)d_in[1];
    const float* in_proj_w  = (const float*)d_in[2];
    const float* in_proj_b  = (const float*)d_in[3];
    const float* weights    = (const float*)d_in[4];
    const float* meas_w     = (const float*)d_in[5];
    const float* out_proj_w = (const float*)d_in[6];
    const float* out_proj_b = (const float*)d_in[7];
    _Float16* ws = (_Float16*)d_ws;
    float* out = (float*)d_out;

    hipLaunchKernelGGL(qprep_kernel, dim3(1), dim3(256), 0, stream,
                       in_proj_w, weights, meas_w, out_proj_w, ws);
    hipLaunchKernelGGL(qconv_main_kernel, dim3(1024), dim3(256), 0, stream,
                       x, inp_scale, in_proj_b, out_proj_b, ws, out);
}
